// RecGINE_56075093017193
// MI455X (gfx1250) — matrix operand order repaired and checked by reading, not compiled, not run on hardware
//
#include <hip/hip_runtime.h>
#include <hip/hip_bf16.h>

// ---------------------------------------------------------------------------
// RecGINE on MI455X (gfx1250, wave32, WMMA).
//  - All GEMMs: v_wmma_f32_16x16x32_bf16, fragments staged in LDS in native
//    WMMA layout so each operand load is 2x ds_load_b128 (no u16 gathers).
//  - Intermediates h/e/t stored bf16 in global (halves edge-phase traffic;
//    they are bf16-WMMA products anyway). Aggregation stays fp32 via
//    global_atomic_add_f32.
//  - Roofline: edge phase ~0.8 GB/layer -> ~110us/3 layers at 23.3 TB/s;
//    GEMM FLOPs (~27 GF) negligible on the WMMA path.
// ---------------------------------------------------------------------------

typedef __attribute__((ext_vector_type(16))) __bf16 v16bf;
typedef __attribute__((ext_vector_type(8)))  float  v8f;

#define N_NODES 50000
#define N_EDGES 800000
#define IN_CH   256
#define HID     128
#define OUT_CH  64

__device__ __forceinline__ unsigned int f32_to_bf16_rne(float f) {
    unsigned int u = __float_as_uint(f);
    u += 0x7FFFu + ((u >> 16) & 1u);        // round-to-nearest-even
    return u >> 16;
}
__device__ __forceinline__ float bf_lo(unsigned int u) { return __uint_as_float(u << 16); }
__device__ __forceinline__ float bf_hi(unsigned int u) { return __uint_as_float(u & 0xFFFF0000u); }

// ---------------------------------------------------------------------------
// C[M,N] = act(A[M,K] @ W[K,N] + bias[N]);  N = NTILES*16, K % 32 == 0.
// 128 threads = 4 waves; block tile 64 x N; wave w does rows [16w,16w+16).
// LDS holds A/B tiles pre-swizzled into WMMA fragment layout:
//   A elem (r,k): Af[r>>4][ (r&15) | (((k&15)>>3)<<4) ][ ((k>>4)<<2)|((k&7)>>1) ]
//   B elem (k,c): Bf[c>>4][ (c&15) | (((k&15)>>3)<<4) ][ ((k>>4)<<2)|((k&7)>>1) ]
// with even-k bf16 in the low half of each dword (ISA 7.12.2 packing).
// ---------------------------------------------------------------------------
template <int NTILES, int ACT, int A_BF16, int OUT_BF16>
__global__ __launch_bounds__(128) void wmma_gemm_kernel(
    const void* __restrict__ Av, const float* __restrict__ W,
    const float* __restrict__ bias, void* __restrict__ Cv,
    int M, int K)
{
    constexpr int N = NTILES * 16;
    __shared__ __align__(32) unsigned int Af[4][32][8];       // 4 KB
    __shared__ __align__(32) unsigned int Bf[NTILES][32][8];  // NTILES KB

    const int tid  = threadIdx.x;
    const int lane = tid & 31;
    const int wave = tid >> 5;
    const int rowBase = blockIdx.x * 64;
    const int mrow = lane & 15;
    const int hsel = lane >> 4;

    v8f acc[NTILES];
#pragma unroll
    for (int n = 0; n < NTILES; ++n) acc[n] = (v8f){0,0,0,0,0,0,0,0};

    for (int kk = 0; kk < K; kk += 32) {
        // ---- stage A tile (64 rows x 32 k) as packed bf16 pairs ----
#pragma unroll 2
        for (int p = tid; p < 64 * 16; p += 128) {
            int r  = p >> 4;             // tile row
            int kp = p & 15;             // dword (k-pair) within 32
            int k  = kp * 2;
            int gr = rowBase + r;
            unsigned int dword;
            if (A_BF16) {
                const unsigned int* A32 = (const unsigned int*)Av;   // bf16 row-major
                dword = (gr < M) ? A32[(size_t)gr * (K >> 1) + (kk >> 1) + kp] : 0u;
            } else {
                const float* A32 = (const float*)Av;
                float2 v = (gr < M) ? *(const float2*)(A32 + (size_t)gr * K + kk + k)
                                    : make_float2(0.0f, 0.0f);
                dword = f32_to_bf16_rne(v.x) | (f32_to_bf16_rne(v.y) << 16);
            }
            int hs = (k & 15) >> 3;
            int dw = ((k >> 4) << 2) | ((k & 7) >> 1);
            Af[r >> 4][(r & 15) | (hs << 4)][dw] = dword;
        }
        // ---- stage W tile (32 k x N) as packed bf16 pairs along k ----
#pragma unroll 2
        for (int q = tid; q < 16 * N; q += 128) {
            int c = q & (N - 1);
            int r = (q / N) * 2;
            float v0 = W[(kk + r) * N + c];
            float v1 = W[(kk + r + 1) * N + c];
            unsigned int dword = f32_to_bf16_rne(v0) | (f32_to_bf16_rne(v1) << 16);
            Bf[c >> 4][(c & 15) | (((r & 15) >> 3) << 4)][((r >> 4) << 2) | ((r & 7) >> 1)] = dword;
        }
        __syncthreads();

        // ---- contiguous 32B fragment loads (2x ds_load_b128 each) ----
        v16bf af = *reinterpret_cast<const v16bf*>(&Af[wave][lane][0]);
#pragma unroll
        for (int n = 0; n < NTILES; ++n) {
            v16bf bfr = *reinterpret_cast<const v16bf*>(&Bf[n][lane][0]);
            acc[n] = __builtin_amdgcn_wmma_f32_16x16x32_bf16(
                false, af, false, bfr, (short)0, acc[n], false, false);
        }
        __syncthreads();
    }

    // ---- epilogue: C/D layout -> VGPR i is row (i + 8*hsel), col = lane%16
#pragma unroll
    for (int n = 0; n < NTILES; ++n) {
        const int col = n * 16 + mrow;
        const float bv = bias[col];
#pragma unroll
        for (int i = 0; i < 8; ++i) {
            int r = rowBase + wave * 16 + i + hsel * 8;
            if (r < M) {
                float v = acc[n][i] + bv;
                if (ACT == 1) v = fmaxf(v, 0.0f);
                if (OUT_BF16)
                    ((unsigned short*)Cv)[(size_t)r * N + col] = (unsigned short)f32_to_bf16_rne(v);
                else
                    ((float*)Cv)[(size_t)r * N + col] = v;
            }
        }
    }
}

// ---------------------------------------------------------------------------
// agg(f32) <- h(bf16): seeds z = h + segment_sum. 4 elems/thread.
// ---------------------------------------------------------------------------
__global__ void seed_agg_kernel(float4* __restrict__ dst,
                                const uint2* __restrict__ src, int n4)
{
    int i = blockIdx.x * blockDim.x + threadIdx.x;
    if (i < n4) {
        uint2 u = src[i];
        float4 o;
        o.x = bf_lo(u.x); o.y = bf_hi(u.x);
        o.z = bf_lo(u.y); o.w = bf_hi(u.y);
        dst[i] = o;
    }
}

// ---------------------------------------------------------------------------
// One wave per edge: msg = relu(h[src] + e); agg[dst] += msg (fp32 atomics).
// bf16 gathers (8B/lane), fp32 global_atomic_add accumulation.
// ---------------------------------------------------------------------------
__global__ __launch_bounds__(256) void edge_scatter_kernel(
    const unsigned short* __restrict__ h, const unsigned short* __restrict__ e,
    const long long* __restrict__ src, const long long* __restrict__ dst,
    float* __restrict__ agg, int E)
{
    int gid  = blockIdx.x * blockDim.x + threadIdx.x;
    int edge = gid >> 5;
    int lane = gid & 31;
    if (edge >= E) return;

    long long s = src[edge];
    long long d = dst[edge];
    int c = lane * 4;

    uint2 hu = *(const uint2*)(h + s * HID + c);
    uint2 eu = *(const uint2*)(e + (long long)edge * HID + c);

    float m0 = fmaxf(bf_lo(hu.x) + bf_lo(eu.x), 0.0f);
    float m1 = fmaxf(bf_hi(hu.x) + bf_hi(eu.x), 0.0f);
    float m2 = fmaxf(bf_lo(hu.y) + bf_lo(eu.y), 0.0f);
    float m3 = fmaxf(bf_hi(hu.y) + bf_hi(eu.y), 0.0f);

    float* dp = agg + d * HID + c;
    unsafeAtomicAdd(dp + 0, m0);   // global_atomic_add_f32
    unsafeAtomicAdd(dp + 1, m1);
    unsafeAtomicAdd(dp + 2, m2);
    unsafeAtomicAdd(dp + 3, m3);
}

// ---------------------------------------------------------------------------
extern "C" void kernel_launch(void* const* d_in, const int* in_sizes, int n_in,
                              void* d_out, int out_size, void* d_ws, size_t ws_size,
                              hipStream_t stream)
{
    (void)in_sizes; (void)n_in; (void)out_size; (void)ws_size;

    const float*      x     = (const float*)d_in[0];
    const long long*  ei    = (const long long*)d_in[1];   // int64 [2, E]
    const float*      ea    = (const float*)d_in[2];
    const float*      W_in  = (const float*)d_in[3];
    const float*      b_in  = (const float*)d_in[4];
    const float*      W_e   = (const float*)d_in[5];
    const float*      b_e   = (const float*)d_in[6];
    const float*      W1    = (const float*)d_in[7];
    const float*      b1    = (const float*)d_in[8];
    const float*      W2    = (const float*)d_in[9];
    const float*      b2    = (const float*)d_in[10];
    const float*      W_out = (const float*)d_in[11];
    const float*      b_out = (const float*)d_in[12];
    float*            out   = (float*)d_out;

    const long long* src = ei;            // edge_index[0]
    const long long* dst = ei + N_EDGES;  // edge_index[1]

    // workspace layout: bf16 intermediates + fp32 aggregation buffer
    unsigned short* e_bf = (unsigned short*)d_ws;               // [E,HID] 204.8 MB
    unsigned short* h_bf = e_bf + (size_t)N_EDGES * HID;        // [N,HID]  12.8 MB
    unsigned short* t_bf = h_bf + (size_t)N_NODES * HID;        // [N,HID]  12.8 MB
    float*          agg  = (float*)(t_bf + (size_t)N_NODES * HID); // [N,HID] 25.6 MB

    const int nodeBlocks = (N_NODES + 63) / 64;      // 782
    const int edgeBlocks = (N_EDGES + 63) / 64;      // 12500
    const int seedN4     = N_NODES * HID / 4;
    const int seedBlocks = (seedN4 + 255) / 256;
    const int scatBlocks = (N_EDGES * 32 + 255) / 256;

    // h = x @ W_in + b_in          (fp32 A, bf16 out, no act)
    wmma_gemm_kernel<8, 0, 0, 1><<<nodeBlocks, 128, 0, stream>>>(x, W_in, b_in, h_bf, N_NODES, IN_CH);
    // e = edge_attr @ W_e + b_e    (layer-invariant, computed once)
    wmma_gemm_kernel<8, 0, 0, 1><<<edgeBlocks, 128, 0, stream>>>(ea, W_e, b_e, e_bf, N_EDGES, 64);

    for (int layer = 0; layer < 3; ++layer) {
        // agg = h (f32)
        seed_agg_kernel<<<seedBlocks, 256, 0, stream>>>((float4*)agg, (const uint2*)h_bf, seedN4);
        // agg += segment_sum(relu(h[src] + e), dst)
        edge_scatter_kernel<<<scatBlocks, 256, 0, stream>>>(h_bf, e_bf, src, dst, agg, N_EDGES);
        // t = relu(agg @ W1 + b1)  (fp32 A, bf16 out)
        wmma_gemm_kernel<8, 1, 0, 1><<<nodeBlocks, 128, 0, stream>>>(agg, W1, b1, t_bf, N_NODES, HID);
        // h = relu(t @ W2 + b2)    (bf16 A: pure swizzle-copy staging)
        wmma_gemm_kernel<8, 1, 1, 1><<<nodeBlocks, 128, 0, stream>>>(t_bf, W2, b2, h_bf, N_NODES, HID);
    }

    // out = h @ W_out + b_out      (bf16 A, fp32 out)
    wmma_gemm_kernel<4, 0, 1, 0><<<nodeBlocks, 128, 0, stream>>>(h_bf, W_out, b_out, out, N_NODES, HID);
}
